// LanguageGenerator_62130996904134
// MI455X (gfx1250) — matrix operand order repaired and checked by reading, not compiled, not run on hardware
//
#include <hip/hip_runtime.h>
#include <hip/hip_bf16.h>

// Problem dims
#define Bz   128
#define Tz   64
#define Vz   16000
#define Ez   128
#define Cz   256
#define Hz   512
#define G4   2048          // 4*H
#define KC   896           // E + C + H  (fused [inp|h] K-dim)
#define MT   32            // decoder row tile per block

#if __has_builtin(__builtin_amdgcn_global_load_async_to_lds_b128)
#define ASYNC_LDS 1
#else
#define ASYNC_LDS 0
#endif

typedef __bf16        v16bf __attribute__((ext_vector_type(16)));
typedef float         v8f   __attribute__((ext_vector_type(8)));
typedef unsigned int  u32x4 __attribute__((ext_vector_type(4)));
typedef int           v4i   __attribute__((ext_vector_type(4)));
// address-space-qualified v4i pointers for the async-to-LDS builtin
typedef __attribute__((address_space(1))) v4i* gv4i_p;   // global
typedef __attribute__((address_space(3))) v4i* lv4i_p;   // LDS

union Frag {
  v16bf bf;
  unsigned short us[16];
  u32x4 q[2];
};

__device__ __forceinline__ unsigned short f2bf(float f) {
  unsigned int u = __builtin_bit_cast(unsigned int, f);
  u += 0x7FFFu + ((u >> 16) & 1u);          // round-to-nearest-even
  return (unsigned short)(u >> 16);
}
__device__ __forceinline__ float sigm(float x) { return 1.0f / (1.0f + __expf(-x)); }

// ---------------- prep kernels ----------------

// zero cell state, zero h-region of A_cat for t=0, zero loss
__global__ void init_kernel(float* c, unsigned short* acat0, float* loss) {
  int i = blockIdx.x * blockDim.x + threadIdx.x;
  if (i < Bz * Hz) {
    c[i] = 0.f;
    acat0[(i >> 9) * KC + (Ez + Cz) + (i & 511)] = 0;
  }
  if (i < Bz) loss[i] = 0.f;
}

// W_cat[j][k] = k<384 ? W_ih[j][k] : W_hh[j][k-384]   (bf16)
__global__ void wcat_kernel(const float* __restrict__ Wih, const float* __restrict__ Whh,
                            unsigned short* __restrict__ Wc) {
  int i = blockIdx.x * blockDim.x + threadIdx.x;
  if (i >= G4 * KC) return;
  int j = i / KC, k = i - j * KC;
  float v = (k < Ez + Cz) ? Wih[j * (Ez + Cz) + k] : Whh[j * Hz + (k - (Ez + Cz))];
  Wc[i] = f2bf(v);
}

__global__ void wdec_kernel(const float* __restrict__ Wdec, unsigned short* __restrict__ Wd) {
  int i = blockIdx.x * blockDim.x + threadIdx.x;
  if (i < Vz * Hz) Wd[i] = f2bf(Wdec[i]);
}

__global__ void bias_kernel(const float* __restrict__ bih, const float* __restrict__ bhh,
                            float* __restrict__ bias) {
  int i = blockIdx.x * blockDim.x + threadIdx.x;
  if (i < G4) bias[i] = bih[i] + bhh[i];
}

// A_cat[t*B+b][0:384] = [ emb_table[x[b][t]] | context[b] ]  (bf16)
__global__ void embed_kernel(const int* __restrict__ x, const float* __restrict__ ctx,
                             const float* __restrict__ emb, unsigned short* __restrict__ Acat) {
  int i = blockIdx.x * blockDim.x + threadIdx.x;
  if (i >= Tz * Bz * (Ez + Cz)) return;
  int row = i / (Ez + Cz);
  int col = i - row * (Ez + Cz);
  int t = row >> 7, b = row & 127;       // row = t*B + b
  float v = (col < Ez) ? emb[x[b * Tz + t] * Ez + col]
                       : ctx[b * Cz + (col - Ez)];
  Acat[row * KC + col] = f2bf(v);
}

// ---------------- recurrent step GEMM: G = A_cat_t @ W_cat^T + bias ----------------
// M=128, N=2048, K=896; one 16x16 tile per wave; grid=128 blocks x 8 waves
__global__ __launch_bounds__(256) void gemm_step_kernel(
    const unsigned short* __restrict__ A,   // [128][896] bf16
    const unsigned short* __restrict__ Bw,  // [2048][896] bf16
    const float* __restrict__ bias,         // [2048]
    float* __restrict__ Gout) {             // [128][2048]
  int wave = threadIdx.x >> 5, lane = threadIdx.x & 31;
  int tile = blockIdx.x * 8 + wave;         // 0..1023
  int m0 = (tile & 7) * 16;                 // M tiles: 8
  int n0 = (tile >> 3) * 16;                // N tiles: 128
  int lm = lane & 15, hi = lane >> 4, lo = hi * 8;

  const unsigned short* arow = A  + (m0 + lm) * KC;   // lane = M row for A frag
  const unsigned short* brow = Bw + (n0 + lm) * KC;   // lane = N col for B frag

  v8f acc = {};
  for (int k0 = 0; k0 < KC; k0 += 32) {
    Frag fa, fb;
    // A 16x32 bf16 layout: VGPR0-3 K=lo..lo+7, VGPR4-7 K=16+lo..16+lo+7
    fa.q[0] = *(const u32x4*)(arow + k0 + lo);
    fa.q[1] = *(const u32x4*)(arow + k0 + 16 + lo);
    // B 32x16 bf16 layout: VGPR0-3 K=lo..lo+7, VGPR4-7 K=16+lo..16+lo+7 (same as A)
    fb.q[0] = *(const u32x4*)(brow + k0 + lo);
    fb.q[1] = *(const u32x4*)(brow + k0 + 16 + lo);
    acc = __builtin_amdgcn_wmma_f32_16x16x32_bf16(false, fa.bf, false, fb.bf,
                                                  (short)0, acc, false, false);
  }
#pragma unroll
  for (int r = 0; r < 8; ++r) {
    int m = m0 + r + hi * 8;               // C/D layout: VGPR r -> M=r (lanes<16), M=8+r
    int n = n0 + lm;
    Gout[m * G4 + n] = acc[r] + bias[n];
  }
}

// ---------------- LSTM gates ----------------
__global__ void gate_kernel(const float* __restrict__ G, float* __restrict__ Cst,
                            unsigned short* __restrict__ acat_next,  // may be null
                            unsigned short* __restrict__ hs, int t) {
  int i = blockIdx.x * blockDim.x + threadIdx.x;
  if (i >= Bz * Hz) return;
  int b = i >> 9, j = i & 511;
  const float* gr = G + b * G4;
  float ig = sigm(gr[j]);
  float fg = sigm(gr[Hz + j]);
  float gg = tanhf(gr[2 * Hz + j]);
  float og = sigm(gr[3 * Hz + j]);
  float c = fg * Cst[i] + ig * gg;
  Cst[i] = c;
  float h = og * tanhf(c);
  unsigned short hb = f2bf(h);
  hs[(b * Tz + t) * Hz + j] = hb;                          // decoder A rows: r = b*T + t
  if (acat_next) acat_next[b * KC + (Ez + Cz) + j] = hb;   // feed next step
}

// ---------------- fused decoder GEMM + online log-softmax + NLL ----------------
// 32 rows per block; 8 waves x 16 cols x 2 M-subtiles -> 128 logit cols/iter, 125 iters.
// Each B fragment is reused by 2 WMMAs; softmax reduction parallel across all 256 threads.
__global__ __launch_bounds__(256) void decoder_loss_kernel(
    const unsigned short* __restrict__ hs,   // [8192][512] bf16, row = b*T+t
    const unsigned short* __restrict__ Wd,   // [16000][512] bf16
    const float* __restrict__ bdec,          // [16000]
    const int* __restrict__ y,               // [8192] flat (b,t)
    float* __restrict__ loss) {              // [128]
  __shared__ unsigned short sA[MT * Hz];     // 32 KB bf16 A tile
  __shared__ float sTile[MT * 128];          // 16 KB logit tile
  __shared__ float sPM[MT * 8];              // per-segment partial max
  __shared__ float sPS[MT * 8];              // per-segment partial sumexp
  __shared__ float sLoss[MT];

  int tid = threadIdx.x;
  int m0 = blockIdx.x * MT;

  // stage A tile (contiguous rows): cache -> LDS directly via async DMA path
#if ASYNC_LDS
  {
    const unsigned short* gsrc = hs + (size_t)m0 * Hz;
    for (int i = tid; i < MT * Hz / 8; i += 256) {
      __builtin_amdgcn_global_load_async_to_lds_b128(
          (gv4i_p)(gsrc + i * 8), (lv4i_p)(sA + i * 8),
          /*imm offset*/ 0, /*cpol*/ 0);
    }
#if __has_builtin(__builtin_amdgcn_s_wait_asynccnt)
    __builtin_amdgcn_s_wait_asynccnt(0);
#else
    asm volatile("s_wait_asynccnt 0x0" ::: "memory");
#endif
  }
#else
  {
    u32x4* dst = (u32x4*)sA;
    const u32x4* src = (const u32x4*)(hs + (size_t)m0 * Hz);
    for (int i = tid; i < MT * Hz / 8; i += 256) dst[i] = src[i];
  }
#endif

  // reduction role: row handled by 8 threads, 16 columns each
  int rrow = tid >> 3, rseg = tid & 7;
  float rm = -3.0e38f, rs = 0.f, rp = 0.f;
  int targ = y[m0 + rrow];
  __syncthreads();

  int wave = tid >> 5, lane = tid & 31;
  int lm = lane & 15, hi = lane >> 4, lo = hi * 8;

  for (int n0 = 0; n0 < Vz; n0 += 128) {
    int nn = n0 + wave * 16 + lm;
    const unsigned short* brow = Wd + (size_t)nn * Hz;
    if (n0 + 128 < Vz)
      __builtin_prefetch(Wd + (size_t)(nn + 128) * Hz, 0, 1);   // global_prefetch_b8

    v8f acc[2] = {};
    for (int k0 = 0; k0 < Hz; k0 += 32) {
      Frag fb;
      fb.q[0] = *(const u32x4*)(brow + k0 + lo);                // global_load_b128
      fb.q[1] = *(const u32x4*)(brow + k0 + 16 + lo);
#pragma unroll
      for (int mi = 0; mi < 2; ++mi) {                          // B reuse x2
        Frag fa;
        const unsigned short* arow = sA + (mi * 16 + lm) * Hz;
        fa.q[0] = *(const u32x4*)(arow + k0 + lo);              // ds_load_b128
        fa.q[1] = *(const u32x4*)(arow + k0 + 16 + lo);
        acc[mi] = __builtin_amdgcn_wmma_f32_16x16x32_bf16(false, fa.bf, false, fb.bf,
                                                          (short)0, acc[mi], false, false);
      }
    }
#pragma unroll
    for (int mi = 0; mi < 2; ++mi) {
#pragma unroll
      for (int r = 0; r < 8; ++r) {
        int m = mi * 16 + r + hi * 8;
        int n = wave * 16 + lm;
        sTile[m * 128 + n] = acc[mi][r] + bdec[n0 + n];
      }
    }
    __syncthreads();

    // parallel partial (max, sumexp): 8 segments x 16 cols per row
    {
      const float* rowv = sTile + rrow * 128 + rseg * 16;
      float pmax = rowv[0];
#pragma unroll
      for (int c2 = 1; c2 < 16; ++c2) pmax = fmaxf(pmax, rowv[c2]);
      float psum = 0.f;
#pragma unroll
      for (int c2 = 0; c2 < 16; ++c2) psum += __expf(rowv[c2] - pmax);
      sPM[rrow * 8 + rseg] = pmax;
      sPS[rrow * 8 + rseg] = psum;
    }
    __syncthreads();

    if (rseg == 0) {                       // online merge for row `rrow`
      float nm = rm;
#pragma unroll
      for (int i2 = 0; i2 < 8; ++i2) nm = fmaxf(nm, sPM[rrow * 8 + i2]);
      float add = 0.f;
#pragma unroll
      for (int i2 = 0; i2 < 8; ++i2)
        add += sPS[rrow * 8 + i2] * __expf(sPM[rrow * 8 + i2] - nm);
      rs = rs * __expf(rm - nm) + add;
      rm = nm;
      int d = targ - n0;
      if (d >= 0 && d < 128) rp = sTile[rrow * 128 + d];
    }
    __syncthreads();                       // protect sTile/sPM/sPS for next iter
  }

  if (rseg == 0) sLoss[rrow] = rp - rm - __logf(rs);
  __syncthreads();
  if (tid == 0) {
    float tot = 0.f;
    for (int r = 0; r < MT; ++r) tot += sLoss[r];
    int b = m0 >> 6;                       // all MT rows belong to batch element m0/64
    atomicAdd(&loss[b], -tot);
  }
}

// ---------------- launcher ----------------
extern "C" void kernel_launch(void* const* d_in, const int* in_sizes, int n_in,
                              void* d_out, int out_size, void* d_ws, size_t ws_size,
                              hipStream_t stream) {
  (void)in_sizes; (void)n_in; (void)out_size; (void)ws_size;
  const int*   x    = (const int*)d_in[0];
  const int*   y    = (const int*)d_in[1];
  const float* ctx  = (const float*)d_in[2];
  const float* emb  = (const float*)d_in[3];
  const float* Wih  = (const float*)d_in[4];
  const float* Whh  = (const float*)d_in[5];
  const float* bih  = (const float*)d_in[6];
  const float* bhh  = (const float*)d_in[7];
  const float* Wdec = (const float*)d_in[8];
  const float* bdec = (const float*)d_in[9];
  float* loss = (float*)d_out;

  char* ws = (char*)d_ws;
  size_t off = 0;
  auto alloc = [&](size_t bytes) -> void* {
    void* p = ws + off;
    off = (off + bytes + 255) & ~size_t(255);
    return p;
  };
  unsigned short* Acat = (unsigned short*)alloc((size_t)Tz * Bz * KC * 2); // 14.7 MB
  unsigned short* Wcat = (unsigned short*)alloc((size_t)G4 * KC * 2);      //  3.7 MB
  unsigned short* Wd   = (unsigned short*)alloc((size_t)Vz * Hz * 2);      // 16.4 MB
  float*          bias = (float*)alloc((size_t)G4 * 4);
  float*          G    = (float*)alloc((size_t)Bz * G4 * 4);               //  1.0 MB
  float*          Cst  = (float*)alloc((size_t)Bz * Hz * 4);
  unsigned short* hs   = (unsigned short*)alloc((size_t)Bz * Tz * Hz * 2); //  8.4 MB

  init_kernel<<<(Bz * Hz + 255) / 256, 256, 0, stream>>>(Cst, Acat, loss);
  wcat_kernel<<<(G4 * KC + 255) / 256, 256, 0, stream>>>(Wih, Whh, Wcat);
  wdec_kernel<<<(Vz * Hz + 255) / 256, 256, 0, stream>>>(Wdec, Wd);
  bias_kernel<<<(G4 + 255) / 256, 256, 0, stream>>>(bih, bhh, bias);
  embed_kernel<<<(Tz * Bz * (Ez + Cz) + 255) / 256, 256, 0, stream>>>(x, ctx, emb, Acat);

  for (int t = 0; t < Tz; ++t) {
    gemm_step_kernel<<<128, 256, 0, stream>>>(Acat + (size_t)t * Bz * KC, Wcat, bias, G);
    gate_kernel<<<(Bz * Hz + 255) / 256, 256, 0, stream>>>(
        G, Cst, (t < Tz - 1) ? (Acat + (size_t)(t + 1) * Bz * KC) : nullptr, hs, t);
  }

  decoder_loss_kernel<<<(Bz * Tz) / MT, 256, 0, stream>>>(hs, Wd, bdec, y, loss);
}
